// SelfAttention3D_60258391163370
// MI455X (gfx1250) — matrix operand order repaired and checked by reading, compiled, not run on hardware
//
#include <hip/hip_runtime.h>
#include <hip/hip_bf16.h>

typedef __attribute__((ext_vector_type(16))) _Float16 v16h;
typedef __attribute__((ext_vector_type(8)))  float    v8f;

#define C_DIM  256
#define N_DIM  4096
#define QKV_ROWS 768
#define HEADS  8
#define DH     32
#define NTILES (N_DIM / 16)   // 256
#define KCHUNK 8              // 256 / 32 reduction chunks

// A-layout K-index map for 16-bit 16x32 A matrices (contiguous: K = i + 16*hb)
__device__ __forceinline__ int map_a(int i, int hb) { return i + 16 * hb; }

__device__ __forceinline__ v8f wmma16(v16h a, v16h b, v8f c) {
  return __builtin_amdgcn_wmma_f32_16x16x32_f16(false, a, false, b, (short)0, c, false, false);
}

// ---------------- 1. GroupNorm statistics (8 groups, 131072 elems each) ----
__global__ void gn_stats(const float* __restrict__ x, float* __restrict__ stats) {
  const int g = blockIdx.x;
  const float4* p = (const float4*)(x + (size_t)g * 131072);
  float s = 0.f, s2 = 0.f;
  for (int i = threadIdx.x; i < 131072 / 4; i += blockDim.x) {
    float4 v = p[i];
    s  += v.x + v.y + v.z + v.w;
    s2 += v.x * v.x + v.y * v.y + v.z * v.z + v.w * v.w;
  }
  __shared__ float sh[512];
  sh[threadIdx.x] = s; sh[256 + threadIdx.x] = s2;
  __syncthreads();
  for (int off = 128; off > 0; off >>= 1) {
    if ((int)threadIdx.x < off) {
      sh[threadIdx.x]       += sh[threadIdx.x + off];
      sh[256 + threadIdx.x] += sh[256 + threadIdx.x + off];
    }
    __syncthreads();
  }
  if (threadIdx.x == 0) {
    float mu  = sh[0] / 131072.f;
    float var = sh[256] / 131072.f - mu * mu;
    stats[g]     = mu;
    stats[8 + g] = rsqrtf(var + 1e-5f);
  }
}

// ---------------- 2. normalize + pack xn into WMMA B-layout f16 ------------
// B-layout tile (cc,it): lane -> N = it*16 + (lane&15); half i -> K = cc*32 + i + 16*(lane>=16)
__global__ void gn_pack(const float* __restrict__ x, const float* __restrict__ gamma,
                        const float* __restrict__ beta, const float* __restrict__ stats,
                        _Float16* __restrict__ xnB) {
  int t = blockIdx.x * blockDim.x + threadIdx.x;          // 0 .. 65535
  int lane = t & 31, tile = t >> 5;
  int cc = tile >> 8, it = tile & 255;
  int hb = lane >> 4, nl = lane & 15;
  int n = it * 16 + nl;
  float mu = stats[cc], rs = stats[8 + cc];               // group == cc (32 channels)
  _Float16* dst = xnB + (size_t)t * 16;
#pragma unroll
  for (int i = 0; i < 16; ++i) {
    int c = cc * 32 + i + 16 * hb;
    float v = (x[(size_t)c * N_DIM + n] - mu) * rs * gamma[c] + beta[c];
    dst[i] = (_Float16)v;
  }
}

// ---------------- 3. pack weights into WMMA A-layout f16 -------------------
__global__ void pack_w(const float* __restrict__ wqkv, const float* __restrict__ wout,
                       _Float16* __restrict__ wA, _Float16* __restrict__ woA) {
  int t = blockIdx.x * blockDim.x + threadIdx.x;          // 0 .. 16383
  int lane = t & 31, tile = t >> 5;
  int hb = lane >> 4, ml = lane & 15;
  if (tile < 48 * 8) {                                    // w_qkv: 48 M-tiles x 8 K-chunks
    int ot = tile >> 3, cc = tile & 7;
    _Float16* dst = wA + (size_t)t * 16;
#pragma unroll
    for (int i = 0; i < 16; ++i)
      dst[i] = (_Float16)wqkv[(size_t)(ot * 16 + ml) * C_DIM + cc * 32 + map_a(i, hb)];
  } else {                                                // w_out: 16 x 8
    int tl = tile - 384;
    int ot = tl >> 3, cc = tl & 7;
    _Float16* dst = woA + ((size_t)tl * 32 + lane) * 16;
#pragma unroll
    for (int i = 0; i < 16; ++i)
      dst[i] = (_Float16)wout[(size_t)(ot * 16 + ml) * C_DIM + cc * 32 + map_a(i, hb)];
  }
}

// ---------------- 4. QKV GEMM: qkv[768x4096] = Wqkv * xn + b ---------------
__global__ void gemm_qkv(const _Float16* __restrict__ wA, const _Float16* __restrict__ xnB,
                         const float* __restrict__ bqkv, _Float16* __restrict__ qkv16) {
  int wv = threadIdx.x >> 5, lane = threadIdx.x & 31;
  int task = blockIdx.x * (blockDim.x >> 5) + wv;         // 0 .. 12287
  int ot = task >> 8, it = task & 255;
  v8f acc = {};
#pragma unroll
  for (int cc = 0; cc < KCHUNK; ++cc) {
    v16h a = *(const v16h*)(wA  + ((size_t)(ot * 8 + cc) * 32 + lane) * 16);
    v16h b = *(const v16h*)(xnB + ((size_t)(cc * NTILES + it) * 32 + lane) * 16);
    acc = wmma16(a, b, acc);
  }
  int hb = lane >> 4, nl = lane & 15;
  int n = it * 16 + nl;
#pragma unroll
  for (int r = 0; r < 8; ++r) {
    int o = ot * 16 + r + 8 * hb;
    qkv16[(size_t)o * N_DIM + n] = (_Float16)(acc[r] + bqkv[o]);
  }
}

// ---------------- 5. pack q (B-layout, pre-scaled), k (A), v^T (A) ---------
__global__ void pack_qkv(const _Float16* __restrict__ qkv16, _Float16* __restrict__ qB,
                         _Float16* __restrict__ kA, _Float16* __restrict__ vA) {
  int t = blockIdx.x * blockDim.x + threadIdx.x;          // 0 .. 196607
  int region = t >> 16;
  int idx = t & 65535;
  int lane = idx & 31, tile = idx >> 5;
  int hb = lane >> 4, ml = lane & 15;
  const float scale = 0.17677669529663687f;               // 1/sqrt(32)
  if (region == 0) {                                      // Q -> B-layout, scaled
    int h = tile >> 8, qt = tile & 255;
    _Float16* dst = qB + (size_t)idx * 16;
#pragma unroll
    for (int i = 0; i < 16; ++i) {
      int d = i + 16 * hb;
      float v = (float)qkv16[(size_t)(h * DH + d) * N_DIM + qt * 16 + ml] * scale;
      dst[i] = (_Float16)v;
    }
  } else if (region == 1) {                               // K -> A-layout (M=key, K=d)
    int h = tile >> 8, jt = tile & 255;
    _Float16* dst = kA + (size_t)idx * 16;
#pragma unroll
    for (int i = 0; i < 16; ++i) {
      int d = map_a(i, hb);
      dst[i] = qkv16[(size_t)(256 + h * DH + d) * N_DIM + jt * 16 + ml];
    }
  } else {                                                // V^T -> A-layout (M=d, K=key)
    int h = tile >> 8, rem = tile & 255;
    int jc = rem >> 1, dt = rem & 1;
    int d = dt * 16 + ml;
    _Float16* dst = vA + (size_t)idx * 16;
#pragma unroll
    for (int i = 0; i < 16; ++i) {
      int j = jc * 32 + map_a(i, hb);
      dst[i] = qkv16[(size_t)(512 + h * DH + d) * N_DIM + j];
    }
  }
}

// ---------------- 6. flash attention: one wave per (head, 16-query tile) ---
__global__ void attn(const _Float16* __restrict__ qB, const _Float16* __restrict__ kA,
                     const _Float16* __restrict__ vA, _Float16* __restrict__ ao16) {
  int wv = threadIdx.x >> 5, lane = threadIdx.x & 31;
  int task = blockIdx.x * (blockDim.x >> 5) + wv;         // 0 .. 2047
  int h = task >> 8, qt = task & 255;
  const bool hi = lane >= 16;
  const int nl = lane & 15;

  v16h b_q = *(const v16h*)(qB + ((size_t)(h * NTILES + qt) * 32 + lane) * 16);
  v8f acc0 = {}, acc1 = {};
  float m = -3.0e38f, l = 0.f;

  for (int jc = 0; jc < 128; ++jc) {
    v16h a_k0 = *(const v16h*)(kA + ((size_t)(h * NTILES + 2 * jc)     * 32 + lane) * 16);
    v16h a_k1 = *(const v16h*)(kA + ((size_t)(h * NTILES + 2 * jc + 1) * 32 + lane) * 16);
    v8f z = {};
    v8f s0 = wmma16(a_k0, b_q, z);                        // S^T keys 32jc+0..15  x queries
    v8f s1 = wmma16(a_k1, b_q, z);                        // S^T keys 32jc+16..31 x queries

    float cm = -3.0e38f;
#pragma unroll
    for (int r = 0; r < 8; ++r) cm = fmaxf(cm, fmaxf(s0[r], s1[r]));
    cm = fmaxf(cm, __shfl_xor(cm, 16, 32));               // merge key halves -> column max
    float mn = fmaxf(m, cm);
    float corr = __expf(m - mn);

    float p0[8], p1[8];
    float rs = 0.f;
#pragma unroll
    for (int r = 0; r < 8; ++r) {
      p0[r] = __expf(s0[r] - mn);
      p1[r] = __expf(s1[r] - mn);
      rs += p0[r] + p1[r];
    }
    rs += __shfl_xor(rs, 16, 32);
    l = l * corr + rs;
    m = mn;

    // realign P^T (D-layout) -> B-layout via lane^16 exchange
    float o0[8], o1[8];
#pragma unroll
    for (int r = 0; r < 8; ++r) { o0[r] = __shfl_xor(p0[r], 16, 32); o1[r] = __shfl_xor(p1[r], 16, 32); }
    v16h bp;
#pragma unroll
    for (int i = 0; i < 8; ++i) {
      bp[i]     = (_Float16)(hi ? o1[i] : p0[i]);
      bp[i + 8] = (_Float16)(hi ? p1[i] : o0[i]);
    }
#pragma unroll
    for (int r = 0; r < 8; ++r) { acc0[r] *= corr; acc1[r] *= corr; }

    v16h a_v0 = *(const v16h*)(vA + ((size_t)((h * 128 + jc) * 2 + 0) * 32 + lane) * 16);
    v16h a_v1 = *(const v16h*)(vA + ((size_t)((h * 128 + jc) * 2 + 1) * 32 + lane) * 16);
    acc0 = wmma16(a_v0, bp, acc0);                        // O^T[d0..15][q]
    acc1 = wmma16(a_v1, bp, acc1);                        // O^T[d16..31][q]
  }

  float inv = 1.0f / l;
  int n = qt * 16 + nl;
#pragma unroll
  for (int r = 0; r < 8; ++r) {
    int d0 = r + 8 * (hi ? 1 : 0);
    ao16[(size_t)(h * DH + d0)      * N_DIM + n] = (_Float16)(acc0[r] * inv);
    ao16[(size_t)(h * DH + 16 + d0) * N_DIM + n] = (_Float16)(acc1[r] * inv);
  }
}

// ---------------- 7. pack attention output into B-layout -------------------
__global__ void pack_ao(const _Float16* __restrict__ ao16, _Float16* __restrict__ aoB) {
  int t = blockIdx.x * blockDim.x + threadIdx.x;          // 0 .. 65535
  int lane = t & 31, tile = t >> 5;
  int cc = tile >> 8, it = tile & 255;
  int hb = lane >> 4, nl = lane & 15;
  _Float16* dst = aoB + (size_t)t * 16;
#pragma unroll
  for (int i = 0; i < 16; ++i)
    dst[i] = ao16[(size_t)(cc * 32 + i + 16 * hb) * N_DIM + it * 16 + nl];
}

// ---------------- 8. output GEMM + bias + residual -------------------------
__global__ void gemm_out(const _Float16* __restrict__ woA, const _Float16* __restrict__ aoB,
                         const float* __restrict__ bout, const float* __restrict__ x,
                         float* __restrict__ out) {
  int wv = threadIdx.x >> 5, lane = threadIdx.x & 31;
  int task = blockIdx.x * (blockDim.x >> 5) + wv;         // 0 .. 4095
  int ot = task >> 8, it = task & 255;
  v8f acc = {};
#pragma unroll
  for (int cc = 0; cc < KCHUNK; ++cc) {
    v16h a = *(const v16h*)(woA + ((size_t)(ot * 8 + cc) * 32 + lane) * 16);
    v16h b = *(const v16h*)(aoB + ((size_t)(cc * NTILES + it) * 32 + lane) * 16);
    acc = wmma16(a, b, acc);
  }
  int hb = lane >> 4, nl = lane & 15;
  int n = it * 16 + nl;
#pragma unroll
  for (int r = 0; r < 8; ++r) {
    int o = ot * 16 + r + 8 * hb;
    size_t off = (size_t)o * N_DIM + n;
    out[off] = acc[r] + bout[o] + x[off];
  }
}

// ---------------------------------------------------------------------------
extern "C" void kernel_launch(void* const* d_in, const int* in_sizes, int n_in,
                              void* d_out, int out_size, void* d_ws, size_t ws_size,
                              hipStream_t stream) {
  const float* x     = (const float*)d_in[0];
  const float* gamma = (const float*)d_in[1];
  const float* beta  = (const float*)d_in[2];
  const float* wqkv  = (const float*)d_in[3];
  const float* bqkv  = (const float*)d_in[4];
  const float* wout  = (const float*)d_in[5];
  const float* bout  = (const float*)d_in[6];
  float* out = (float*)d_out;

  char* p = (char*)d_ws;
  float* stats = (float*)p;                 p += 256;
  _Float16* xnB   = (_Float16*)p;           p += (size_t)C_DIM * N_DIM * 2;        // 2 MB
  _Float16* wA    = (_Float16*)p;           p += (size_t)QKV_ROWS * C_DIM * 2;     // 384 KB
  _Float16* woA   = (_Float16*)p;           p += (size_t)C_DIM * C_DIM * 2;        // 128 KB
  _Float16* qkv16 = (_Float16*)p;           p += (size_t)QKV_ROWS * N_DIM * 2;     // 6 MB
  _Float16* qB    = (_Float16*)p;           p += (size_t)C_DIM * N_DIM * 2;
  _Float16* kA    = (_Float16*)p;           p += (size_t)C_DIM * N_DIM * 2;
  _Float16* vA    = (_Float16*)p;           p += (size_t)C_DIM * N_DIM * 2;
  _Float16* ao16  = (_Float16*)p;           p += (size_t)C_DIM * N_DIM * 2;
  _Float16* aoB   = (_Float16*)p;           p += (size_t)C_DIM * N_DIM * 2;

  gn_stats<<<8, 256, 0, stream>>>(x, stats);
  gn_pack <<<256, 256, 0, stream>>>(x, gamma, beta, stats, xnB);
  pack_w  <<<64, 256, 0, stream>>>(wqkv, wout, wA, woA);
  gemm_qkv<<<1536, 256, 0, stream>>>(wA, xnB, bqkv, qkv16);   // 12288 tile-waves
  pack_qkv<<<768, 256, 0, stream>>>(qkv16, qB, kA, vA);
  attn    <<<256, 256, 0, stream>>>(qB, kA, vA, ao16);        // 2048 (head,qtile) waves
  pack_ao <<<256, 256, 0, stream>>>(ao16, aoB);
  gemm_out<<<512, 256, 0, stream>>>(woA, aoB, bout, x, out);  // 4096 tile-waves
}
